// OffSetAttention_56564719289017
// MI455X (gfx1250) — matrix operand order repaired and checked by reading, not compiled, not run on hardware
//
#include <hip/hip_runtime.h>
#include <hip/hip_bf16.h>
#include <math.h>

// Problem shape (fixed by reference): x[B,C,N], wq[D,C], wv/wt[C,C]
#define BB 4
#define CC 256
#define NN 4096
#define DD 64

typedef __bf16 bf16x16 __attribute__((ext_vector_type(16)));
typedef float  f32x8   __attribute__((ext_vector_type(8)));
typedef int    v4i     __attribute__((ext_vector_type(4)));

#define AS1 __attribute__((address_space(1)))
#define AS3 __attribute__((address_space(3)))

#if __has_builtin(__builtin_amdgcn_global_load_async_to_lds_b128) && \
    __has_builtin(__builtin_amdgcn_s_wait_asynccnt)
#define ASYNC_LDS 1
#else
#define ASYNC_LDS 0
#endif

struct __align__(16) U4 { unsigned x, y, z, w; };
union FragAB { bf16x16 v; U4 q[2]; };

__device__ __forceinline__ unsigned short f2bf(float f) {
    unsigned u = __float_as_uint(f);
    u += 0x7FFFu + ((u >> 16) & 1u);       // round-to-nearest-even
    return (unsigned short)(u >> 16);
}

// 16-byte global->LDS copy: async on CDNA5 (ASYNCcnt), fallback = regular ld/st.
__device__ __forceinline__ void cp16_g2s(const unsigned short* g, unsigned short* s) {
#if ASYNC_LDS
    __builtin_amdgcn_global_load_async_to_lds_b128((AS1 v4i*)g, (AS3 v4i*)s, 0, 0);
#else
    *(U4*)s = *(const U4*)g;
#endif
}
__device__ __forceinline__ void async_wait_all() {
#if ASYNC_LDS
    __builtin_amdgcn_s_wait_asynccnt(0);
#endif
}

// A-fragment: 16x32 bf16 tile, src row-major [16 x ld], K contiguous.
// lane<16: row=lane, elems 0-7 = K 0..7, 8-15 = K 16..23 ; lane>=16: K +8.
__device__ __forceinline__ bf16x16 load_a_frag(const unsigned short* src, int ld) {
    const int lane = threadIdx.x & 31;
    const int row  = lane & 15;
    const int kb   = (lane >> 4) << 3;      // 0 or 8
    const unsigned short* p = src + (size_t)row * ld + kb;
    FragAB f;
    f.q[0] = *(const U4*)(p);               // K = kb .. kb+7
    f.q[1] = *(const U4*)(p + 16);          // K = kb+16 .. kb+23
    return f.v;
}

// B-fragment: 32x16 bf16 tile; src stored TRANSPOSED as [16 cols x ld], K contiguous.
__device__ __forceinline__ bf16x16 load_b_frag(const unsigned short* src, int ld) {
    const int lane = threadIdx.x & 31;
    const int col  = lane & 15;
    const int kb   = (lane >> 4) << 3;      // 0 or 8
    const unsigned short* p = src + (size_t)col * ld + kb;
    FragAB f;
    f.q[0] = *(const U4*)(p);
    f.q[1] = *(const U4*)(p + 16);
    return f.v;
}

__device__ __forceinline__ f32x8 wmma_bf16(bf16x16 a, bf16x16 b, f32x8 c) {
    return __builtin_amdgcn_wmma_f32_16x16x32_bf16(false, a, false, b, (short)0, c,
                                                   false, false);
}

__device__ __forceinline__ U4 pack8(const f32x8 a) {
    U4 o;
    o.x = (unsigned)f2bf(a[0]) | ((unsigned)f2bf(a[1]) << 16);
    o.y = (unsigned)f2bf(a[2]) | ((unsigned)f2bf(a[3]) << 16);
    o.z = (unsigned)f2bf(a[4]) | ((unsigned)f2bf(a[5]) << 16);
    o.w = (unsigned)f2bf(a[6]) | ((unsigned)f2bf(a[7]) << 16);
    return o;
}

// ---------------------------------------------------------------- converts
__global__ void cvt_bf16_kernel(const float* __restrict__ in,
                                unsigned short* __restrict__ out, int n) {
    int i = blockIdx.x * blockDim.x + threadIdx.x;
    if (i < n) out[i] = f2bf(in[i]);
}

// xT[b][n][c] = bf16(x[b][c][n])  (C contiguous, for B-fragments of projections)
__global__ void xpose_kernel(const float* __restrict__ x,
                             unsigned short* __restrict__ xT) {
    size_t idx = (size_t)blockIdx.x * blockDim.x + threadIdx.x;
    int n = (int)(idx % NN);
    int c = (int)((idx / NN) % CC);
    int b = (int)(idx / ((size_t)NN * CC));
    xT[((size_t)b * NN + n) * CC + c] = f2bf(x[idx]);
}

// Stage a shared 16 x CC weight strip into LDS (one-shot, 128 threads).
__device__ __forceinline__ void stage_w_strip(const unsigned short* wBase,
                                              unsigned short* sW) {
#pragma unroll
    for (int i = 0; i < 4; ++i) {                 // 16*256/8 = 512 xfers / 128 thr
        int li  = threadIdx.x + i * 128;
        int row = li >> 5;                        // 32 xfers per row
        int ko  = (li & 31) * 8;
        cp16_g2s(wBase + (size_t)row * CC + ko, sW + row * CC + ko);
    }
    async_wait_all();
    __syncthreads();
}

// ------------------------------------------------- K projection: Kt[b][n][d]
// Kt = (wq @ x)^T stored d-contiguous: serves both A- and B-fragments of energy.
__global__ void proj_k_kernel(const unsigned short* __restrict__ wqb,
                              const unsigned short* __restrict__ xT,
                              unsigned short* __restrict__ Kt) {
    __shared__ __align__(16) unsigned short sW[16 * CC];
    const int wid = threadIdx.x >> 5;
    const int nt  = blockIdx.x * 4 + wid;   // n tile
    const int dt  = blockIdx.y;             // d tile
    const int b   = blockIdx.z;
    stage_w_strip(wqb + (size_t)dt * 16 * CC, sW);
    const unsigned short* xTb = xT + (size_t)b * NN * CC;
    f32x8 acc = {0.f,0.f,0.f,0.f,0.f,0.f,0.f,0.f};
#pragma unroll
    for (int k = 0; k < CC; k += 32) {
        bf16x16 af = load_a_frag(sW + k, CC);
        bf16x16 bf = load_b_frag(xTb + (size_t)nt * 16 * CC + k, CC);
        acc = wmma_bf16(af, bf, acc);
    }
    const int lane = threadIdx.x & 31;
    const int n    = nt * 16 + (lane & 15);
    const int dof  = dt * 16 + ((lane >> 4) << 3);
    *(U4*)(Kt + ((size_t)b * NN + n) * DD + dof) = pack8(acc);
}

// ------------------------------------------------- V projection: xvb[b][c][n]
__global__ void proj_v_kernel(const unsigned short* __restrict__ wvb,
                              const unsigned short* __restrict__ xT,
                              const float* __restrict__ bv,
                              unsigned short* __restrict__ xvb) {
    __shared__ __align__(16) unsigned short sW[16 * CC];
    const int wid = threadIdx.x >> 5;
    const int nt  = blockIdx.x * 4 + wid;
    const int ct  = blockIdx.y;
    const int b   = blockIdx.z;
    stage_w_strip(wvb + (size_t)ct * 16 * CC, sW);
    const unsigned short* xTb = xT + (size_t)b * NN * CC;
    f32x8 acc = {0.f,0.f,0.f,0.f,0.f,0.f,0.f,0.f};
#pragma unroll
    for (int k = 0; k < CC; k += 32) {
        bf16x16 af = load_a_frag(sW + k, CC);
        bf16x16 bf = load_b_frag(xTb + (size_t)nt * 16 * CC + k, CC);
        acc = wmma_bf16(af, bf, acc);
    }
    const int lane  = threadIdx.x & 31;
    const int n     = nt * 16 + (lane & 15);
    const int cbase = ct * 16 + ((lane >> 4) << 3);
#pragma unroll
    for (int r = 0; r < 8; ++r)
        xvb[((size_t)b * CC + cbase + r) * NN + n] = f2bf(acc[r] + bv[cbase + r]);
}

// --------------------------------- row stats of energy = Kt Kt^T (Gram matrix)
__global__ void row_stats_kernel(const unsigned short* __restrict__ Kt,
                                 float* __restrict__ rowmax,
                                 float* __restrict__ rowsum) {
    const int n0 = blockIdx.x * 16;
    const int b  = blockIdx.y;
    const unsigned short* K = Kt + (size_t)b * NN * DD;
    const bf16x16 a0 = load_a_frag(K + (size_t)n0 * DD, DD);
    const bf16x16 a1 = load_a_frag(K + (size_t)n0 * DD + 32, DD);
    f32x8 vmax = {-1e30f,-1e30f,-1e30f,-1e30f,-1e30f,-1e30f,-1e30f,-1e30f};
    for (int m0 = 0; m0 < NN; m0 += 16) {
        bf16x16 b0 = load_b_frag(K + (size_t)m0 * DD, DD);
        bf16x16 b1 = load_b_frag(K + (size_t)m0 * DD + 32, DD);
        f32x8 e = {0.f,0.f,0.f,0.f,0.f,0.f,0.f,0.f};
        e = wmma_bf16(a0, b0, e);
        e = wmma_bf16(a1, b1, e);
#pragma unroll
        for (int r = 0; r < 8; ++r) vmax[r] = fmaxf(vmax[r], e[r]);
    }
#pragma unroll
    for (int off = 1; off <= 8; off <<= 1)
#pragma unroll
        for (int r = 0; r < 8; ++r)
            vmax[r] = fmaxf(vmax[r], __shfl_xor(vmax[r], off, 32));
    const int lane  = threadIdx.x & 31;
    const int rbase = (lane >> 4) << 3;
    if ((lane & 15) == 0) {
#pragma unroll
        for (int r = 0; r < 8; ++r)
            rowmax[(size_t)b * NN + n0 + rbase + r] = vmax[r];
    }
    f32x8 vsum = {0.f,0.f,0.f,0.f,0.f,0.f,0.f,0.f};
    for (int m0 = 0; m0 < NN; m0 += 16) {
        bf16x16 b0 = load_b_frag(K + (size_t)m0 * DD, DD);
        bf16x16 b1 = load_b_frag(K + (size_t)m0 * DD + 32, DD);
        f32x8 e = {0.f,0.f,0.f,0.f,0.f,0.f,0.f,0.f};
        e = wmma_bf16(a0, b0, e);
        e = wmma_bf16(a1, b1, e);
#pragma unroll
        for (int r = 0; r < 8; ++r) vsum[r] += __expf(e[r] - vmax[r]);
    }
#pragma unroll
    for (int off = 1; off <= 8; off <<= 1)
#pragma unroll
        for (int r = 0; r < 8; ++r) vsum[r] += __shfl_xor(vsum[r], off, 32);
    if ((lane & 15) == 0) {
#pragma unroll
        for (int r = 0; r < 8; ++r)
            rowsum[(size_t)b * NN + n0 + rbase + r] = vsum[r];
    }
}

// --------------------------------- column sums of softmax(attn) over n
__global__ void col_sums_kernel(const unsigned short* __restrict__ Kt,
                                const float* __restrict__ rowmax,
                                const float* __restrict__ rowsum,
                                float* __restrict__ colsum) {
    const int m0 = blockIdx.x * 16;
    const int b  = blockIdx.y;
    const unsigned short* K = Kt + (size_t)b * NN * DD;
    const bf16x16 b0 = load_b_frag(K + (size_t)m0 * DD, DD);
    const bf16x16 b1 = load_b_frag(K + (size_t)m0 * DD + 32, DD);
    const int lane  = threadIdx.x & 31;
    const int rbase = (lane >> 4) << 3;
    float csum = 0.f;
    for (int n0 = 0; n0 < NN; n0 += 16) {
        bf16x16 a0 = load_a_frag(K + (size_t)n0 * DD, DD);
        bf16x16 a1 = load_a_frag(K + (size_t)n0 * DD + 32, DD);
        f32x8 e = {0.f,0.f,0.f,0.f,0.f,0.f,0.f,0.f};
        e = wmma_bf16(a0, b0, e);
        e = wmma_bf16(a1, b1, e);
        const float* rm = rowmax + (size_t)b * NN + n0 + rbase;
        const float* rs = rowsum + (size_t)b * NN + n0 + rbase;
#pragma unroll
        for (int r = 0; r < 8; ++r) csum += __expf(e[r] - rm[r]) / rs[r];
    }
    csum += __shfl_xor(csum, 16, 32);
    if (lane < 16) colsum[(size_t)b * NN + m0 + lane] = csum;
}

// ------------- materialize normalized attention, bf16, TRANSPOSED: attnT[m][n]
__global__ void attn_kernel(const unsigned short* __restrict__ Kt,
                            const float* __restrict__ rowmax,
                            const float* __restrict__ rowsum,
                            const float* __restrict__ colsum,
                            unsigned short* __restrict__ attnT, int b) {
    const int wid = threadIdx.x >> 5;
    const int mt  = blockIdx.x * 4 + wid;
    const int nt  = blockIdx.y;
    const unsigned short* K = Kt + (size_t)b * NN * DD;
    bf16x16 a0 = load_a_frag(K + (size_t)nt * 16 * DD, DD);
    bf16x16 a1 = load_a_frag(K + (size_t)nt * 16 * DD + 32, DD);
    bf16x16 b0 = load_b_frag(K + (size_t)mt * 16 * DD, DD);
    bf16x16 b1 = load_b_frag(K + (size_t)mt * 16 * DD + 32, DD);
    f32x8 e = {0.f,0.f,0.f,0.f,0.f,0.f,0.f,0.f};
    e = wmma_bf16(a0, b0, e);
    e = wmma_bf16(a1, b1, e);
    const int lane  = threadIdx.x & 31;
    const int rbase = (lane >> 4) << 3;
    const float cs = 1.f / (1e-9f + colsum[(size_t)b * NN + mt * 16 + (lane & 15)]);
    const float* rm = rowmax + (size_t)b * NN + nt * 16 + rbase;
    const float* rs = rowsum + (size_t)b * NN + nt * 16 + rbase;
    f32x8 at;
#pragma unroll
    for (int r = 0; r < 8; ++r) at[r] = __expf(e[r] - rm[r]) / rs[r] * cs;
    *(U4*)(attnT + ((size_t)mt * 16 + (lane & 15)) * NN + nt * 16 + rbase) = pack8(at);
}

// ------------- x_a = x_v @ attn, fused x_d = x - x_a, stored bf16 xdT[b][n][c]
// Shared x_v channel strip double-buffered in LDS via async global->LDS copies.
#define KCH 128                               // k-chunk (n) per stage
__global__ void xa_xd_kernel(const unsigned short* __restrict__ xvb,
                             const unsigned short* __restrict__ attnT,
                             const float* __restrict__ x,
                             unsigned short* __restrict__ xdT, int b) {
    __shared__ __align__(16) unsigned short sA[2][16 * KCH];
    const int wid = threadIdx.x >> 5;
    const int mt  = blockIdx.x * 4 + wid;     // output n tile (per wave)
    const int ct  = blockIdx.y;               // channel tile (shared by block)
    const unsigned short* aBase = xvb + ((size_t)b * CC + ct * 16) * NN;
    const unsigned short* bBase = attnT + (size_t)mt * 16 * NN;

    auto stage = [&](int chunk, int buf) {    // 16 x KCH ushorts = 256 x 16B
#pragma unroll
        for (int i = 0; i < (16 * KCH / 8) / 128; ++i) {
            int li  = threadIdx.x + i * 128;
            int row = li >> 4;                // KCH/8 = 16 xfers per row
            int ko  = (li & 15) * 8;
            cp16_g2s(aBase + (size_t)row * NN + chunk * KCH + ko,
                     &sA[buf][row * KCH + ko]);
        }
    };

    f32x8 acc = {0.f,0.f,0.f,0.f,0.f,0.f,0.f,0.f};
    stage(0, 0);
    const int NCH = NN / KCH;                 // 32 chunks
    for (int kb = 0; kb < NCH; ++kb) {
        async_wait_all();                     // s_wait_asynccnt 0
        __syncthreads();                      // buffer kb visible to all waves
        if (kb + 1 < NCH) stage(kb + 1, (kb + 1) & 1);
        const unsigned short* sa = &sA[kb & 1][0];
        __builtin_prefetch(bBase + (kb + 1) * KCH, 0, 3);   // global_prefetch_b8
#pragma unroll
        for (int k4 = 0; k4 < KCH; k4 += 32) {
            bf16x16 af = load_a_frag(sa + k4, KCH);               // ds_load_b128
            bf16x16 bf = load_b_frag(bBase + kb * KCH + k4, NN);  // global (L2)
            acc = wmma_bf16(af, bf, acc);
        }
    }
    const int lane  = threadIdx.x & 31;
    const int n     = mt * 16 + (lane & 15);
    const int cbase = ct * 16 + ((lane >> 4) << 3);
    f32x8 xd;
#pragma unroll
    for (int r = 0; r < 8; ++r)
        xd[r] = x[((size_t)b * CC + cbase + r) * NN + n] - acc[r];
    *(U4*)(xdT + ((size_t)b * NN + n) * CC + cbase) = pack8(xd);
}

// ------------------------- t = wt @ x_d + bt  -> fp32 tbuf[b][c][n]
__global__ void tconv_kernel(const unsigned short* __restrict__ wtb,
                             const unsigned short* __restrict__ xdT,
                             const float* __restrict__ bt,
                             float* __restrict__ tbuf) {
    __shared__ __align__(16) unsigned short sW[16 * CC];
    const int wid = threadIdx.x >> 5;
    const int nt  = blockIdx.x * 4 + wid;
    const int ot  = blockIdx.y;
    const int b   = blockIdx.z;
    stage_w_strip(wtb + (size_t)ot * 16 * CC, sW);
    const unsigned short* bBase = xdT + ((size_t)b * NN + nt * 16) * CC;
    f32x8 acc = {0.f,0.f,0.f,0.f,0.f,0.f,0.f,0.f};
#pragma unroll
    for (int k = 0; k < CC; k += 32) {
        bf16x16 af = load_a_frag(sW + k, CC);
        bf16x16 bf = load_b_frag(bBase + k, CC);
        acc = wmma_bf16(af, bf, acc);
    }
    const int lane  = threadIdx.x & 31;
    const int n     = nt * 16 + (lane & 15);
    const int obase = ot * 16 + ((lane >> 4) << 3);
#pragma unroll
    for (int r = 0; r < 8; ++r)
        tbuf[((size_t)b * CC + obase + r) * NN + n] = acc[r] + bt[obase + r];
}

// ------------------------- BatchNorm batch stats per channel (LDS reduction)
__global__ void bn_stats_kernel(const float* __restrict__ tbuf,
                                float* __restrict__ mean,
                                float* __restrict__ rstd) {
    const int c = blockIdx.x;
    __shared__ float s1[256];
    __shared__ float s2[256];
    float s = 0.f, q = 0.f;
    for (int i = threadIdx.x; i < BB * NN; i += blockDim.x) {
        int b = i / NN, n = i - b * NN;
        float v = tbuf[((size_t)b * CC + c) * NN + n];
        s += v; q += v * v;
    }
    s1[threadIdx.x] = s; s2[threadIdx.x] = q;
    __syncthreads();
    for (int st = 128; st > 0; st >>= 1) {
        if (threadIdx.x < st) {
            s1[threadIdx.x] += s1[threadIdx.x + st];
            s2[threadIdx.x] += s2[threadIdx.x + st];
        }
        __syncthreads();
    }
    if (threadIdx.x == 0) {
        const float inv = 1.f / (float)(BB * NN);
        float m = s1[0] * inv;
        float var = s2[0] * inv - m * m;
        mean[c] = m;
        rstd[c] = rsqrtf(var + 1e-5f);
    }
}

// ------------------------- out = x + relu(gamma * t_hat + beta)
__global__ void final_kernel(const float* __restrict__ x,
                             const float* __restrict__ tbuf,
                             const float* __restrict__ mean,
                             const float* __restrict__ rstd,
                             const float* __restrict__ gamma,
                             const float* __restrict__ beta,
                             float* __restrict__ out) {
    size_t idx = (size_t)blockIdx.x * blockDim.x + threadIdx.x;
    int c = (int)((idx / NN) % CC);
    float th = (tbuf[idx] - mean[c]) * rstd[c];
    float v = gamma[c] * th + beta[c];
    out[idx] = x[idx] + fmaxf(v, 0.f);
}

// ======================================================================
extern "C" void kernel_launch(void* const* d_in, const int* in_sizes, int n_in,
                              void* d_out, int out_size, void* d_ws, size_t ws_size,
                              hipStream_t stream) {
    const float* x     = (const float*)d_in[0];
    const float* wq    = (const float*)d_in[1];
    const float* wv    = (const float*)d_in[2];
    const float* bv    = (const float*)d_in[3];
    const float* wt    = (const float*)d_in[4];
    const float* bt    = (const float*)d_in[5];
    const float* gamma = (const float*)d_in[6];
    const float* beta  = (const float*)d_in[7];
    float* out = (float*)d_out;

    char*  ws  = (char*)d_ws;
    size_t off = 0;
    auto carve = [&](size_t bytes) -> void* {
        off = (off + 255) & ~(size_t)255;
        void* p = ws + off;
        off += bytes;
        return p;
    };
    unsigned short* wqb    = (unsigned short*)carve((size_t)DD * CC * 2);
    unsigned short* wvb    = (unsigned short*)carve((size_t)CC * CC * 2);
    unsigned short* wtb    = (unsigned short*)carve((size_t)CC * CC * 2);
    unsigned short* xT     = (unsigned short*)carve((size_t)BB * NN * CC * 2);
    unsigned short* Kt     = (unsigned short*)carve((size_t)BB * NN * DD * 2);
    unsigned short* xvb    = (unsigned short*)carve((size_t)BB * CC * NN * 2);
    unsigned short* attnT  = (unsigned short*)carve((size_t)NN * NN * 2);   // per-batch
    unsigned short* xdT    = (unsigned short*)carve((size_t)BB * NN * CC * 2);
    float*          tbuf   = (float*)carve((size_t)BB * CC * NN * 4);
    float*          rowmax = (float*)carve((size_t)BB * NN * 4);
    float*          rowsum = (float*)carve((size_t)BB * NN * 4);
    float*          colsum = (float*)carve((size_t)BB * NN * 4);
    float*          meanb  = (float*)carve((size_t)CC * 4);
    float*          rstdb  = (float*)carve((size_t)CC * 4);
    (void)ws_size; (void)n_in; (void)in_sizes; (void)out_size;

    // weights + activations -> bf16
    cvt_bf16_kernel<<<(DD * CC + 255) / 256, 256, 0, stream>>>(wq, wqb, DD * CC);
    cvt_bf16_kernel<<<(CC * CC + 255) / 256, 256, 0, stream>>>(wv, wvb, CC * CC);
    cvt_bf16_kernel<<<(CC * CC + 255) / 256, 256, 0, stream>>>(wt, wtb, CC * CC);
    xpose_kernel<<<(BB * CC * NN) / 256, 256, 0, stream>>>(x, xT);

    // projections (WMMA, weight strip staged in LDS via async copies)
    proj_k_kernel<<<dim3(NN / 64, DD / 16, BB), 128, 0, stream>>>(wqb, xT, Kt);
    proj_v_kernel<<<dim3(NN / 64, CC / 16, BB), 128, 0, stream>>>(wvb, xT, bv, xvb);

    // softmax row stats + column-L1 stats (WMMA-recomputed Gram energy)
    row_stats_kernel<<<dim3(NN / 16, BB), 32, 0, stream>>>(Kt, rowmax, rowsum);
    col_sums_kernel<<<dim3(NN / 16, BB), 32, 0, stream>>>(Kt, rowmax, rowsum, colsum);

    // per-batch: bf16 attn (L2-resident) then x_a GEMM fused with x_d
    for (int b = 0; b < BB; ++b) {
        attn_kernel<<<dim3(NN / 64, NN / 16), 128, 0, stream>>>(
            Kt, rowmax, rowsum, colsum, attnT, b);
        xa_xd_kernel<<<dim3(NN / 64, CC / 16), 128, 0, stream>>>(
            xvb, attnT, x, xdT, b);
    }

    // trans_conv GEMM, BN stats, residual output
    tconv_kernel<<<dim3(NN / 64, CC / 16, BB), 128, 0, stream>>>(wtb, xdT, bt, tbuf);
    bn_stats_kernel<<<CC, 256, 0, stream>>>(tbuf, meanb, rstdb);
    final_kernel<<<(BB * CC * NN) / 256, 256, 0, stream>>>(
        x, tbuf, meanb, rstdb, gamma, beta, out);
}
